// IDXST_IDCT_1133871366825
// MI455X (gfx1250) — matrix operand order repaired and checked by reading, not compiled, not run on hardware
//
#include <hip/hip_runtime.h>

// Out = S @ (X @ C1): two 4096^3 GEMMs, run as split-bf16 "bf16x3"
// (A ~ Ah+Al, B ~ Bh+Bl; acc += Ah*Bh + Ah*Bl + Al*Bh in fp32) on
// V_WMMA_F32_16X16X32_BF16 — ~8x MACs/instr vs the fp32 WMMA shape,
// 3x emulation cost => ~2.5x fewer matrix-pipe issues, ~1e-5 rel error.

typedef __attribute__((ext_vector_type(4)))  unsigned int v4u;
typedef __attribute__((ext_vector_type(4)))  float v4f;
typedef __attribute__((ext_vector_type(8)))  float v8f;
typedef __attribute__((ext_vector_type(8)))  __bf16 v8bf;
typedef __attribute__((ext_vector_type(16))) __bf16 v16bf;

#define NDIM 4096
#define PI_OVER_8192 3.83495196971410292009e-4f  // pi/8192

// RNE split of fp32 into hi/lo bf16 halves.
__device__ __forceinline__ void split2(float x, unsigned short& h,
                                       unsigned short& l) {
  unsigned u = __float_as_uint(x);
  unsigned rh = (u + 0x7FFFu + ((u >> 16) & 1u)) & 0xFFFF0000u;
  float lo = x - __uint_as_float(rh);
  unsigned ul = __float_as_uint(lo);
  h = (unsigned short)(rh >> 16);
  l = (unsigned short)((ul + 0x7FFFu + ((ul >> 16) & 1u)) >> 16);
}

// ---------------------------------------------------------------------------
// X (fp32 row-major) -> Xh/Xl (bf16 row-major). 8 elements/thread.
// ---------------------------------------------------------------------------
__global__ __launch_bounds__(256) void convert_x_split(
    const float* __restrict__ X, unsigned short* __restrict__ Xh,
    unsigned short* __restrict__ Xl) {
  size_t idx = (size_t)blockIdx.x * 256 + threadIdx.x;  // over 4096*512
  size_t e0 = idx * 8;
  v4f x0 = *(const v4f*)(X + e0);
  v4f x1 = *(const v4f*)(X + e0 + 4);
  float v[8] = {x0[0], x0[1], x0[2], x0[3], x1[0], x1[1], x1[2], x1[3]};
  unsigned hq[4], lq[4];
#pragma unroll
  for (int j = 0; j < 4; ++j) {
    unsigned short h0, l0, h1, l1;
    split2(v[2 * j], h0, l0);
    split2(v[2 * j + 1], h1, l1);
    hq[j] = (unsigned)h0 | ((unsigned)h1 << 16);
    lq[j] = (unsigned)l0 | ((unsigned)l1 << 16);
  }
  v4u qh = {hq[0], hq[1], hq[2], hq[3]};
  v4u ql = {lq[0], lq[1], lq[2], lq[3]};
  *(v4u*)(Xh + e0) = qh;
  *(v4u*)(Xl + e0) = ql;
}

// ---------------------------------------------------------------------------
// C1[k,u] = cos(pi*k*(2u+1)/2N), split hi/lo, stored 16-row-grouped over k
// (the WMMA bf16 B-fragment feed): elem (k,u) at ((k>>4)*N + u)*16 + (k&15).
// Exact range reduction: k*(2u+1) mod 16384 (power of two, fits u32).
// One thread computes one (k16,u) column chunk = 16 values = one 32B run.
// ---------------------------------------------------------------------------
__global__ __launch_bounds__(256) void gen_cos_split(
    unsigned short* __restrict__ Ch, unsigned short* __restrict__ Cl) {
  size_t idx = (size_t)blockIdx.x * 256 + threadIdx.x;  // k16*4096 + u
  int u = (int)(idx & (NDIM - 1));
  int k16 = (int)(idx >> 12);
  unsigned w = (unsigned)(2 * u + 1);
  unsigned hq[8], lq[8];
#pragma unroll
  for (int j = 0; j < 16; j += 2) {
    unsigned t0 = ((unsigned)(k16 * 16 + j) * w) & 16383u;
    unsigned t1 = ((unsigned)(k16 * 16 + j + 1) * w) & 16383u;
    unsigned short h0, l0, h1, l1;
    split2(cosf((float)t0 * PI_OVER_8192), h0, l0);
    split2(cosf((float)t1 * PI_OVER_8192), h1, l1);
    hq[j >> 1] = (unsigned)h0 | ((unsigned)h1 << 16);
    lq[j >> 1] = (unsigned)l0 | ((unsigned)l1 << 16);
  }
  size_t base = idx * 16;
  v4u qh0 = {hq[0], hq[1], hq[2], hq[3]}, qh1 = {hq[4], hq[5], hq[6], hq[7]};
  v4u ql0 = {lq[0], lq[1], lq[2], lq[3]}, ql1 = {lq[4], lq[5], lq[6], lq[7]};
  ((v4u*)(Ch + base))[0] = qh0;
  ((v4u*)(Ch + base))[1] = qh1;
  ((v4u*)(Cl + base))[0] = ql0;
  ((v4u*)(Cl + base))[1] = ql1;
}

// ---------------------------------------------------------------------------
// S[u,k] = sin(pi*(2u+1)*k/2M), split hi/lo, row-major (A matrix of GEMM2).
// ---------------------------------------------------------------------------
__global__ __launch_bounds__(256) void gen_sin_split(
    unsigned short* __restrict__ Sh, unsigned short* __restrict__ Sl) {
  size_t idx = (size_t)blockIdx.x * 256 + threadIdx.x;  // over 4096*512
  int u = (int)(idx >> 9);
  int k0 = (int)(idx & 511) * 8;
  unsigned w = (unsigned)(2 * u + 1);
  unsigned hq[4], lq[4];
#pragma unroll
  for (int j = 0; j < 8; j += 2) {
    unsigned t0 = ((unsigned)(k0 + j) * w) & 16383u;
    unsigned t1 = ((unsigned)(k0 + j + 1) * w) & 16383u;
    unsigned short h0, l0, h1, l1;
    split2(sinf((float)t0 * PI_OVER_8192), h0, l0);
    split2(sinf((float)t1 * PI_OVER_8192), h1, l1);
    hq[j >> 1] = (unsigned)h0 | ((unsigned)h1 << 16);
    lq[j >> 1] = (unsigned)l0 | ((unsigned)l1 << 16);
  }
  size_t e0 = idx * 8;
  v4u qh = {hq[0], hq[1], hq[2], hq[3]};
  v4u ql = {lq[0], lq[1], lq[2], lq[3]};
  *(v4u*)(Sh + e0) = qh;
  *(v4u*)(Sl + e0) = ql;
}

// ---------------------------------------------------------------------------
// bf16x3 GEMM, all dims 4096.
//   Ah/Al : bf16 row-major
//   Bh/Bl : bf16 16-row-grouped over K (fragment-feed layout)
//   SPLIT_OUT: emit hi/lo bf16 in 16-row-grouped layout (next GEMM's B);
//              else fp32 row-major.
// Block 128x128, BK=32, 8 waves (2x4) of 64x32 -> 24 WMMA/wave/iter.
// ---------------------------------------------------------------------------
#define CAT16(lo, hi) \
  __builtin_shufflevector(lo, hi, 0, 1, 2, 3, 4, 5, 6, 7, 8, 9, 10, 11, 12, 13, 14, 15)

template <bool SPLIT_OUT>
__global__ __launch_bounds__(256) void bf16x3_gemm_4096(
    const unsigned short* __restrict__ Ah, const unsigned short* __restrict__ Al,
    const unsigned short* __restrict__ Bh, const unsigned short* __restrict__ Bl,
    unsigned short* __restrict__ Oh, unsigned short* __restrict__ Ol,
    float* __restrict__ Of) {
  constexpr int N = NDIM;
  constexpr int ASTR = 40;  // bf16/row: 80B rows, conflict-free frag reads
  constexpr int BSTR = 24;  // bf16/col: 48B cols, each half-wave covers 64 banks
  __shared__ alignas(16) unsigned short AsH[128 * ASTR], AsL[128 * ASTR];
  __shared__ alignas(16) unsigned short BsH[256 * BSTR], BsL[256 * BSTR];

  const int tid = threadIdx.x, lane = tid & 31, wave = tid >> 5;
  const int wm = wave >> 2, wn = wave & 3;
  const int m0 = blockIdx.y * 128, n0 = blockIdx.x * 128;

  // staging maps
  const int arow = tid >> 1, ac0 = (tid & 1) * 16;    // A: 2 thr/row, 16 bf16 each
  const int bc16 = tid >> 7, bcol = tid & 127;        // B: one 32B column run each

  const unsigned short* aPh = Ah + (size_t)(m0 + arow) * N + ac0;
  const unsigned short* aPl = Al + (size_t)(m0 + arow) * N + ac0;
  const size_t bOff = ((size_t)bc16 * N + n0 + bcol) * 16;

  v4u sA[4], sB[4];
  auto loadTiles = [&](int kt) {
    const unsigned short* pa = aPh + (size_t)kt * 32;
    const unsigned short* pl = aPl + (size_t)kt * 32;
    sA[0] = *(const v4u*)(pa);
    sA[1] = *(const v4u*)(pa + 8);
    sA[2] = *(const v4u*)(pl);
    sA[3] = *(const v4u*)(pl + 8);
    const size_t bo = bOff + (size_t)kt * (2u * N * 16u);
    sB[0] = *(const v4u*)(Bh + bo);
    sB[1] = *(const v4u*)(Bh + bo + 8);
    sB[2] = *(const v4u*)(Bl + bo);
    sB[3] = *(const v4u*)(Bl + bo + 8);
  };
  auto stageTiles = [&]() {
    const int ai = arow * ASTR + ac0;
    *(v4u*)&AsH[ai] = sA[0];
    *(v4u*)&AsH[ai + 8] = sA[1];
    *(v4u*)&AsL[ai] = sA[2];
    *(v4u*)&AsL[ai + 8] = sA[3];
    const int bi = (bc16 * 128 + bcol) * BSTR;
    *(v4u*)&BsH[bi] = sB[0];
    *(v4u*)&BsH[bi + 8] = sB[1];
    *(v4u*)&BsL[bi] = sB[2];
    *(v4u*)&BsL[bi + 8] = sB[3];
  };

  v8f acc[4][2] = {};
  const int fr = lane & 15, g = lane >> 4;

  loadTiles(0);
  for (int kt = 0; kt < N / 32; ++kt) {
    stageTiles();
    __syncthreads();
    if (kt + 1 < N / 32) loadTiles(kt + 1);  // prefetch across compute

    v16bf aH[4], aL[4], bH[2], bL[2];
#pragma unroll
    for (int mt = 0; mt < 4; ++mt) {
      const int ro = (wm * 64 + mt * 16 + fr) * ASTR + 16 * g;
      aH[mt] = CAT16(*(const v8bf*)&AsH[ro], *(const v8bf*)&AsH[ro + 8]);
      aL[mt] = CAT16(*(const v8bf*)&AsL[ro], *(const v8bf*)&AsL[ro + 8]);
    }
#pragma unroll
    for (int nt = 0; nt < 2; ++nt) {
      const int co = (g * 128 + wn * 32 + nt * 16 + fr) * BSTR;
      bH[nt] = CAT16(*(const v8bf*)&BsH[co], *(const v8bf*)&BsH[co + 8]);
      bL[nt] = CAT16(*(const v8bf*)&BsL[co], *(const v8bf*)&BsL[co + 8]);
    }
#pragma unroll
    for (int mt = 0; mt < 4; ++mt)
#pragma unroll
      for (int nt = 0; nt < 2; ++nt) {
        acc[mt][nt] = __builtin_amdgcn_wmma_f32_16x16x32_bf16(
            false, aH[mt], false, bH[nt], (short)0, acc[mt][nt], false, false);
        acc[mt][nt] = __builtin_amdgcn_wmma_f32_16x16x32_bf16(
            false, aH[mt], false, bL[nt], (short)0, acc[mt][nt], false, false);
        acc[mt][nt] = __builtin_amdgcn_wmma_f32_16x16x32_bf16(
            false, aL[mt], false, bH[nt], (short)0, acc[mt][nt], false, false);
      }
    __syncthreads();
  }

  // epilogue: VGPR j of a tile = (row rb+j, col c); rb>>4 constant over j, so
  // split-output rows pack into one contiguous 16B run per tile per array.
#pragma unroll
  for (int mt = 0; mt < 4; ++mt) {
#pragma unroll
    for (int nt = 0; nt < 2; ++nt) {
      const int rb = m0 + wm * 64 + mt * 16 + 8 * g;
      const int c = n0 + wn * 32 + nt * 16 + fr;
      if (SPLIT_OUT) {
        unsigned hq[4], lq[4];
#pragma unroll
        for (int j = 0; j < 4; ++j) {
          unsigned short h0, l0, h1, l1;
          split2(acc[mt][nt][2 * j], h0, l0);
          split2(acc[mt][nt][2 * j + 1], h1, l1);
          hq[j] = (unsigned)h0 | ((unsigned)h1 << 16);
          lq[j] = (unsigned)l0 | ((unsigned)l1 << 16);
        }
        const size_t base = ((size_t)(rb >> 4) * N + c) * 16 + (rb & 15);
        v4u qh = {hq[0], hq[1], hq[2], hq[3]};
        v4u ql = {lq[0], lq[1], lq[2], lq[3]};
        *(v4u*)(Oh + base) = qh;
        *(v4u*)(Ol + base) = ql;
      } else {
#pragma unroll
        for (int j = 0; j < 8; ++j)
          Of[(size_t)(rb + j) * N + c] = acc[mt][nt][j];
      }
    }
  }
}

// ---------------------------------------------------------------------------
extern "C" void kernel_launch(void* const* d_in, const int* in_sizes, int n_in,
                              void* d_out, int out_size, void* d_ws,
                              size_t ws_size, hipStream_t stream) {
  (void)in_sizes; (void)n_in; (void)out_size; (void)ws_size;
  const float* x = (const float*)d_in[0];  // [4096,4096] fp32
  // expk tables unused: bases regenerated exactly (integer range reduction).

  const size_t M32 = (size_t)32 << 20;  // 32 MB per bf16 matrix
  unsigned short* Xh  = (unsigned short*)d_ws;
  unsigned short* Xl  = (unsigned short*)((char*)d_ws + 1 * M32);
  unsigned short* C1h = (unsigned short*)((char*)d_ws + 2 * M32);
  unsigned short* C1l = (unsigned short*)((char*)d_ws + 3 * M32);
  unsigned short* Sh  = (unsigned short*)((char*)d_ws + 4 * M32);
  unsigned short* Sl  = (unsigned short*)((char*)d_ws + 5 * M32);
  unsigned short* Y1h = (unsigned short*)((char*)d_ws + 6 * M32);
  unsigned short* Y1l = (unsigned short*)((char*)d_ws + 7 * M32);
  float* out = (float*)d_out;

  convert_x_split<<<8192, 256, 0, stream>>>(x, Xh, Xl);
  gen_cos_split<<<4096, 256, 0, stream>>>(C1h, C1l);
  gen_sin_split<<<8192, 256, 0, stream>>>(Sh, Sl);

  dim3 grid(NDIM / 128, NDIM / 128);  // 32 x 32
  // Stage 1: Y1 = X @ C1, emitted pre-split in B-fragment layout
  bf16x3_gemm_4096<true><<<grid, 256, 0, stream>>>(Xh, Xl, C1h, C1l, Y1h, Y1l,
                                                   nullptr);
  // Stage 2: Out = S @ Y1
  bf16x3_gemm_4096<false><<<grid, 256, 0, stream>>>(Sh, Sl, Y1h, Y1l, nullptr,
                                                    nullptr, out);
}
